// GridKNNDownsample_14748917694822
// MI455X (gfx1250) — matrix operand order repaired and checked by reading, not compiled, not run on hardware
//
#include <hip/hip_runtime.h>
#include <hip/hip_bf16.h>

typedef _Float16 h16 __attribute__((ext_vector_type(16)));
typedef _Float16 h8  __attribute__((ext_vector_type(8)));
typedef _Float16 h2  __attribute__((ext_vector_type(2)));
typedef float    f8  __attribute__((ext_vector_type(8)));

namespace {
constexpr int CIN   = 192;   // input channels
constexpr int COUT  = 384;   // output channels
constexpr int KNN   = 16;    // neighbors (== WMMA M dim)
constexpr int KCH   = 6;     // CIN / 32 K-chunks per WMMA reduction
constexpr int NT    = 3;     // N-tiles per wave (8 waves * 3 * 16 = 384 = COUT)
constexpr int MB    = 8;     // m rows processed per block
constexpr int ASTR  = 200;   // padded LDS row stride (f16 elems) -> conflict-free b128
constexpr float EPS = 1e-5f;
}

// --- fp32 -> f16 weight conversion (tiny, runs once per launch) ---
__global__ void wconvert_kernel(const float* __restrict__ W,
                                _Float16* __restrict__ Wh, int n)
{
    int i = blockIdx.x * blockDim.x + threadIdx.x;
    if (i < n) Wh[i] = (_Float16)W[i];
}

__launch_bounds__(256, 1)
__global__ void knn_ln_proj_max_kernel(const float* __restrict__ src,
                                       const int*   __restrict__ idx,
                                       const float* __restrict__ gamma,
                                       const float* __restrict__ beta,
                                       const _Float16* __restrict__ Wh,
                                       float* __restrict__ out)
{
    __shared__ _Float16 As[KNN * ASTR];

    const int tid  = threadIdx.x;
    const int lane = tid & 31;
    const int wv   = tid >> 5;      // wave id 0..7
    const int half = lane >> 4;     // 0 = lanes 0-15, 1 = lanes 16-31
    const int l16  = lane & 15;

    // --------------------------------------------------------------
    // Preload B fragments (weights) into VGPRs; reused for all MB rows.
    // B[k,n] = W[o0+n, c0+k]; W row-major => per lane (n = l16) the 16
    // K-values for this half-wave are 16 contiguous f16 in a W row.
    // --------------------------------------------------------------
    h16 bfrag[NT][KCH];
    #pragma unroll
    for (int t = 0; t < NT; ++t) {
        const int o = wv * (NT * 16) + t * 16 + l16;
        const _Float16* wrow = Wh + (size_t)o * CIN + half * 16;
        #pragma unroll
        for (int k = 0; k < KCH; ++k) {
            const h8* p = (const h8*)(wrow + k * 32);
            h8 lo = p[0];
            h8 hi = p[1];
            bfrag[t][k] = __builtin_shufflevector(lo, hi,
                0,1,2,3,4,5,6,7,8,9,10,11,12,13,14,15);
        }
    }

    const int m0 = blockIdx.x * MB;
    for (int mi = 0; mi < MB; ++mi) {
        const int m = m0 + mi;

        __syncthreads();  // previous iteration's A readers are done

        // ----------------------------------------------------------
        // Gather + LayerNorm: each wave produces 2 of the 16 rows.
        // Lane handles channel pairs {2L, 2L+64, 2L+128}.
        // ----------------------------------------------------------
        #pragma unroll
        for (int rr = 0; rr < 2; ++rr) {
            const int r  = wv * 2 + rr;
            const int nn = idx[(size_t)m * KNN + r];
            const float* srow = src + (size_t)nn * CIN;
            const int c0 = lane * 2;

            float2 x0 = *(const float2*)(srow + c0);
            float2 x1 = *(const float2*)(srow + c0 + 64);
            float2 x2 = *(const float2*)(srow + c0 + 128);

            float s  = x0.x + x0.y + x1.x + x1.y + x2.x + x2.y;
            float sq = x0.x*x0.x + x0.y*x0.y + x1.x*x1.x + x1.y*x1.y
                     + x2.x*x2.x + x2.y*x2.y;
            #pragma unroll
            for (int off = 16; off > 0; off >>= 1) {
                s  += __shfl_xor(s,  off, 32);
                sq += __shfl_xor(sq, off, 32);
            }
            const float mean = s * (1.0f / CIN);
            const float var  = sq * (1.0f / CIN) - mean * mean;
            const float rs   = rsqrtf(var + EPS);

            float2 g0 = *(const float2*)(gamma + c0);
            float2 g1 = *(const float2*)(gamma + c0 + 64);
            float2 g2 = *(const float2*)(gamma + c0 + 128);
            float2 b0 = *(const float2*)(beta  + c0);
            float2 b1 = *(const float2*)(beta  + c0 + 64);
            float2 b2 = *(const float2*)(beta  + c0 + 128);

            h2 y0, y1, y2;
            y0.x = (_Float16)((x0.x - mean) * rs * g0.x + b0.x);
            y0.y = (_Float16)((x0.y - mean) * rs * g0.y + b0.y);
            y1.x = (_Float16)((x1.x - mean) * rs * g1.x + b1.x);
            y1.y = (_Float16)((x1.y - mean) * rs * g1.y + b1.y);
            y2.x = (_Float16)((x2.x - mean) * rs * g2.x + b2.x);
            y2.y = (_Float16)((x2.y - mean) * rs * g2.y + b2.y);

            _Float16* arow = As + r * ASTR;
            *(h2*)(arow + c0)       = y0;
            *(h2*)(arow + c0 + 64)  = y1;
            *(h2*)(arow + c0 + 128) = y2;
        }
        __syncthreads();

        // ----------------------------------------------------------
        // WMMA: A[16 x 192] (LDS) x B tiles (VGPR) -> 3 f32 16x16 tiles.
        // A layout (matches B): lanes 0-15 hold K {0..15},
        //                       lanes 16-31 hold K {16..31}.
        // ----------------------------------------------------------
        f8 acc[NT] = {};
        #pragma unroll
        for (int k = 0; k < KCH; ++k) {
            const h8* pa = (const h8*)(As + l16 * ASTR + k * 32 + half * 16);
            h8 alo = pa[0];        // K base + (half?16:0)     .. +7
            h8 ahi = pa[1];        // +8 f16 further along row
            h16 a = __builtin_shufflevector(alo, ahi,
                0,1,2,3,4,5,6,7,8,9,10,11,12,13,14,15);
            #pragma unroll
            for (int t = 0; t < NT; ++t) {
                acc[t] = __builtin_amdgcn_wmma_f32_16x16x32_f16(
                    /*neg_a=*/false, a, /*neg_b=*/false, bfrag[t][k],
                    /*c_mod=*/(short)0, acc[t],
                    /*reuse_a=*/false, /*reuse_b=*/false);
            }
        }

        // ----------------------------------------------------------
        // Max over the 16 neighbor rows:
        // D layout: VGPR j = row (j + 8*half... rows j and j+8 split
        // across lane halves), col = lane%16.  Reduce the 8 VGPRs, then
        // xor-16 across halves, lanes 0-15 hold the per-column max.
        // ----------------------------------------------------------
        #pragma unroll
        for (int t = 0; t < NT; ++t) {
            float mx = acc[t][0];
            #pragma unroll
            for (int j = 1; j < 8; ++j) mx = fmaxf(mx, acc[t][j]);
            mx = fmaxf(mx, __shfl_xor(mx, 16, 32));
            if (lane < 16) {
                out[(size_t)m * COUT + wv * (NT * 16) + t * 16 + lane] = mx;
            }
        }
    }
}

extern "C" void kernel_launch(void* const* d_in, const int* in_sizes, int n_in,
                              void* d_out, int out_size, void* d_ws, size_t ws_size,
                              hipStream_t stream) {
    const float* src   = (const float*)d_in[0];  // [N, 192] fp32
    const int*   idx   = (const int*)  d_in[1];  // [M, 16]  int32
    const float* gamma = (const float*)d_in[2];  // [192]
    const float* beta  = (const float*)d_in[3];  // [192]
    const float* W     = (const float*)d_in[4];  // [384, 192] fp32
    float*       out   = (float*)d_out;          // [M, 384] fp32

    _Float16* Wh = (_Float16*)d_ws;              // 384*192*2 = 144 KB scratch

    const int nW = COUT * CIN;
    wconvert_kernel<<<(nW + 255) / 256, 256, 0, stream>>>(W, Wh, nW);

    const int M = in_sizes[1] / KNN;             // 32768
    knn_ln_proj_max_kernel<<<M / MB, 256, 0, stream>>>(src, idx, gamma, beta, Wh, out);
}
